// EntropyModule_74354473828491
// MI455X (gfx1250) — matrix operand order repaired and checked by reading, not compiled, not run on hardware
//
#include <hip/hip_runtime.h>
#include <math.h>

// ---------------------------------------------------------------------------
// MI455X (gfx1250) implementation of the graph-attention entropy module.
//  - wave32-centric: one point (16 neighbors == WMMA M=16) per wave
//  - all GEMMs via v_wmma_f32_16x16x32_f16 (f16 in, f32 accumulate)
//  - weights pre-packed into B-fragment layout (two b128 loads / fragment)
//  - gathered neighbor features -> A fragments with two contiguous b128 loads
//  - per-wave LDS scratch for D-tile relayout; no cross-wave sync needed
// ---------------------------------------------------------------------------

typedef __attribute__((ext_vector_type(16))) _Float16 v16h;
typedef __attribute__((ext_vector_type(8)))  float    v8f;
typedef __attribute__((ext_vector_type(4)))  unsigned int u32x4;

union Frag {
  v16h v;
  _Float16 h[16];
  u32x4 q[2];
};

__device__ __forceinline__ v8f wmma_f16(v16h a, v16h b, v8f c) {
  return __builtin_amdgcn_wmma_f32_16x16x32_f16(false, a, false, b,
                                                (short)0, c, false, false);
}

// B fragment: packed as [(chunk*Ntiles + tile)*32 + lane]*16 halves, contiguous.
__device__ __forceinline__ v16h load_frag(const _Float16* p) {
  Frag f;
  f.q[0] = *(const u32x4*)(p);
  f.q[1] = *(const u32x4*)(p + 8);
  return f.v;
}

// A fragment from a contiguous f16 row (gathered neighbor features).
// 16-bit A layout: half h -> K = c*32 + h + 8*(h>=8) + 8*(lane>=16)
// => two contiguous 8-half (16B) runs.
__device__ __forceinline__ v16h frag_row(const _Float16* row, int c, int lane) {
  int b = (lane >= 16) ? 8 : 0;
  Frag f;
  f.q[0] = *(const u32x4*)(row + c * 32 + b);
  f.q[1] = *(const u32x4*)(row + c * 32 + 16 + b);
  return f.v;
}

// A fragment from per-wave LDS f32 buffer (16 rows x ncols), cvt f32->f16.
__device__ __forceinline__ v16h frag_lds(const float* hb, int ncols, int c, int lane) {
  int m = lane & 15;
  int kb = (lane >= 16) ? 8 : 0;
  Frag f;
#pragma unroll
  for (int h = 0; h < 16; ++h) {
    int k = c * 32 + kb + h + ((h >= 8) ? 8 : 0);
    f.h[h] = (_Float16)hb[m * ncols + k];
  }
  return f.v;
}

// D tile (16x16 f32) -> LDS row-major, + bias, optional relu.
// D layout: lane holds (M = r + 8*(lane>=16), N = lane&15) for r=0..7.
__device__ __forceinline__ void store_tile(float* hb, int ncols, int t, v8f acc,
                                           const float* bias, int lane, bool do_relu) {
  int n = lane & 15;
  int mb = (lane >> 4) * 8;
  float bv = bias[t * 16 + n];
#pragma unroll
  for (int r = 0; r < 8; ++r) {
    float v = acc[r] + bv;
    if (do_relu) v = fmaxf(v, 0.0f);
    hb[(mb + r) * ncols + t * 16 + n] = v;
  }
}

// Attention (h @ [aw1 | aw2] via WMMA) + leakyrelu + softmax over K=16 +
// ELU(weighted sum) + optional residual; writes f32 and f16 outputs.
__device__ __forceinline__ void attn_tail(const float* hb, float* st,
                                          const _Float16* ap, float abias, int lane,
                                          const float* resid, float* xoutf,
                                          _Float16* xout16, int point) {
  v8f acc = {};
#pragma unroll
  for (int c = 0; c < 4; ++c) {
    v16h a = frag_lds(hb, 128, c, lane);
    v16h b = load_frag(ap + (c * 32 + lane) * 16);
    acc = wmma_f16(a, b, acc);
  }
  int n = lane & 15, mb = (lane >> 4) * 8;
#pragma unroll
  for (int r = 0; r < 8; ++r) st[(mb + r) * 16 + n] = acc[r];

  // s_k = dot(h_k, aw[0:128]) + dot(h_0, aw[128:256]) + b  (col0 + [0][1])
  float d0 = st[1];
  float s[16], mx = -1e30f;
#pragma unroll
  for (int k = 0; k < 16; ++k) {
    float v = st[k * 16] + d0 + abias;
    v = (v > 0.0f) ? v : 0.2f * v;  // leaky_relu 0.2
    s[k] = v;
    mx = fmaxf(mx, v);
  }
  float sum = 0.0f;
#pragma unroll
  for (int k = 0; k < 16; ++k) { s[k] = __expf(s[k] - mx); sum += s[k]; }
  float inv = 1.0f / sum;

#pragma unroll
  for (int j = 0; j < 4; ++j) {
    int d = lane + j * 32;
    float v = 0.0f;
#pragma unroll
    for (int k = 0; k < 16; ++k) v += s[k] * hb[k * 128 + d];
    v *= inv;
    v = (v > 0.0f) ? v : (__expf(v) - 1.0f);  // elu
    if (resid) v += resid[(size_t)point * 128 + d];
    if (xoutf) xoutf[(size_t)point * 128 + d] = v;
    xout16[(size_t)point * 128 + d] = (_Float16)v;
  }
}

// ---------------------------------------------------------------------------
// Weight packer: row-major f32 W[Kact][Nact] -> f16 B-fragment layout with
// K padded to chunks*32, N to Nt*16. mode==1: attention pack (256x1 weight
// into columns 0/1 of a single 128x16 tile group).
// ---------------------------------------------------------------------------
__global__ void pack_weight_kernel(const float* __restrict__ W, _Float16* __restrict__ dst,
                                   int Kact, int Nact, int Nt, int chunks, int mode,
                                   int total) {
  int e = blockIdx.x * blockDim.x + threadIdx.x;
  if (e >= total) return;
  int h = e & 15;
  int lane = (e >> 4) & 31;
  int rest = e >> 9;
  int t = rest % Nt;
  int c = rest / Nt;
  int K = c * 32 + h + ((h >= 8) ? 8 : 0) + ((lane >= 16) ? 8 : 0);
  int Ncol = t * 16 + (lane & 15);
  float v = 0.0f;
  if (mode == 0) {
    if (K < Kact && Ncol < Nact) v = W[K * Nact + Ncol];
  } else {
    if (K < 128) {
      if (Ncol == 0) v = W[K];
      else if (Ncol == 1) v = W[128 + K];
    }
  }
  dst[e] = (_Float16)v;
}

// ---------------------------------------------------------------------------
// Layer 0: feat[16x12] -> 32 -> 64 -> 128 (relu) + attention head.
// ---------------------------------------------------------------------------
__global__ __launch_bounds__(128) void graph_layer0(
    const float* __restrict__ inp, const int* __restrict__ knn,
    const _Float16* __restrict__ wp1, const float* __restrict__ b1,
    const _Float16* __restrict__ wp2, const float* __restrict__ b2,
    const _Float16* __restrict__ wp3, const float* __restrict__ b3,
    const _Float16* __restrict__ ap, const float* __restrict__ abp,
    float* __restrict__ x0f, _Float16* __restrict__ x016, int N) {
  __shared__ float hb_s[4][16 * 128];
  __shared__ float tb_s[4][16 * 64];
  __shared__ float st_s[4][256];
  int lane = threadIdx.x & 31;
  int wid = threadIdx.x >> 5;
  int point = blockIdx.x * 4 + wid;
  if (point >= N) return;
  float* hb = hb_s[wid];
  float* tb = tb_s[wid];
  float* st = st_s[wid];

  int m = lane & 15;
  int nb = knn[point * 16 + m];
  float pix = inp[3 * point], piy = inp[3 * point + 1], piz = inp[3 * point + 2];
  float pjx = inp[3 * nb], pjy = inp[3 * nb + 1], pjz = inp[3 * nb + 2];
  float sx = pjx - pix, sy = pjy - piy, sz = pjz - piz;
  float f12[12] = {pjx, pjy, pjz, pjx, pjy, pjz,
                   sx,  sy,  sz,  fabsf(sx), fabsf(sy), fabsf(sz)};
  v16h a0;
  {
    int kb = (lane >= 16) ? 8 : 0;
    Frag f;
#pragma unroll
    for (int h = 0; h < 16; ++h) {
      int k = kb + h + ((h >= 8) ? 8 : 0);
      f.h[h] = (k < 12) ? (_Float16)f12[k] : (_Float16)0.0f;
    }
    a0 = f.v;
  }
  // h1 = relu(feat @ W1 + b1): 2 tiles -> hb (ncols 32)
#pragma unroll
  for (int t = 0; t < 2; ++t) {
    v8f acc = {};
    acc = wmma_f16(a0, load_frag(wp1 + (t * 32 + lane) * 16), acc);
    store_tile(hb, 32, t, acc, b1, lane, true);
  }
  // h2 = relu(h1 @ W2 + b2): 4 tiles -> tb (ncols 64)
  {
    v16h a1 = frag_lds(hb, 32, 0, lane);
#pragma unroll
    for (int t = 0; t < 4; ++t) {
      v8f acc = {};
      acc = wmma_f16(a1, load_frag(wp2 + (t * 32 + lane) * 16), acc);
      store_tile(tb, 64, t, acc, b2, lane, true);
    }
  }
  // h3 = relu(h2 @ W3 + b3): 2 chunks x 8 tiles -> hb (ncols 128)
  {
    v16h a2[2];
    a2[0] = frag_lds(tb, 64, 0, lane);
    a2[1] = frag_lds(tb, 64, 1, lane);
#pragma unroll
    for (int t = 0; t < 8; ++t) {
      v8f acc = {};
#pragma unroll
      for (int c = 0; c < 2; ++c)
        acc = wmma_f16(a2[c], load_frag(wp3 + ((c * 8 + t) * 32 + lane) * 16), acc);
      store_tile(hb, 128, t, acc, b3, lane, true);
    }
  }
  attn_tail(hb, st, ap, abp[0], lane, nullptr, x0f, x016, point);
}

// ---------------------------------------------------------------------------
// Layers 1/2: feat[16x137] (gathered x + geometry) -> 128 (relu) + attention,
// residual x_prev. 5 K-chunks x 8 N-tiles of WMMA per point.
// ---------------------------------------------------------------------------
__global__ __launch_bounds__(128) void graph_layer(
    const float* __restrict__ inp, const int* __restrict__ knn,
    const _Float16* __restrict__ xprev16, const float* __restrict__ xprevf,
    const _Float16* __restrict__ wpack, const float* __restrict__ bias,
    const _Float16* __restrict__ apack, const float* __restrict__ abp,
    float* __restrict__ xoutf, _Float16* __restrict__ xout16, int N) {
  __shared__ float hb_s[4][16 * 128];
  __shared__ float st_s[4][256];
  int lane = threadIdx.x & 31;
  int wid = threadIdx.x >> 5;
  int point = blockIdx.x * 4 + wid;
  if (point >= N) return;
  float* hb = hb_s[wid];
  float* st = st_s[wid];

  int m = lane & 15;
  int nb = knn[point * 16 + m];
  float pix = inp[3 * point], piy = inp[3 * point + 1], piz = inp[3 * point + 2];
  float pjx = inp[3 * nb], pjy = inp[3 * nb + 1], pjz = inp[3 * nb + 2];
  float sx = pjx - pix, sy = pjy - piy, sz = pjz - piz;
  float g[9] = {pjx, pjy, pjz, sx, sy, sz, fabsf(sx), fabsf(sy), fabsf(sz)};

  // A fragments: 4 chunks from the gathered f16 feature row + 1 geometry chunk.
  const _Float16* row = xprev16 + (size_t)nb * 128;
  v16h afrag[5];
#pragma unroll
  for (int c = 0; c < 4; ++c) afrag[c] = frag_row(row, c, lane);
  {
    int kb = (lane >= 16) ? 8 : 0;
    Frag f;
#pragma unroll
    for (int h = 0; h < 16; ++h) {
      int kk = kb + h + ((h >= 8) ? 8 : 0);
      f.h[h] = (kk < 9) ? (_Float16)g[kk] : (_Float16)0.0f;
    }
    afrag[4] = f.v;
  }

  // h = relu(feat @ W + b): 8 tiles x 5 chunks.
#pragma unroll
  for (int t = 0; t < 8; ++t) {
    v8f acc = {};
#pragma unroll
    for (int c = 0; c < 5; ++c)
      acc = wmma_f16(afrag[c], load_frag(wpack + ((c * 8 + t) * 32 + lane) * 16), acc);
    store_tile(hb, 128, t, acc, bias, lane, true);
  }
  attn_tail(hb, st, apack, abp[0], lane, xprevf, xoutf, xout16, point);
}

// ---------------------------------------------------------------------------
// Head: f = [x0|x1|x2] (16 points/wave): 384->128 relu, 128->128 relu,
// 128->16, then mu / exp(sigma).
// ---------------------------------------------------------------------------
__global__ __launch_bounds__(128) void head_kernel(
    const _Float16* __restrict__ xh16, const _Float16* __restrict__ po0,
    const float* __restrict__ ob0, const _Float16* __restrict__ po1,
    const float* __restrict__ ob1, const _Float16* __restrict__ po2,
    const float* __restrict__ ob2, float* __restrict__ out, int N) {
  __shared__ float hb_s[4][16 * 128];
  int lane = threadIdx.x & 31;
  int wid = threadIdx.x >> 5;
  int base = (blockIdx.x * 4 + wid) * 16;
  if (base >= N) return;
  float* hb = hb_s[wid];
  int m = lane & 15;
  int p = base + m;
  int kb = (lane >= 16) ? 8 : 0;

  v8f zero = {};
  v8f acc[8];
#pragma unroll
  for (int t = 0; t < 8; ++t) acc[t] = zero;

  // GEMM1: A[16x384] x B[384x128]; 12 chunks x 8 tiles.
#pragma unroll
  for (int c = 0; c < 12; ++c) {
    const _Float16* row = xh16 + ((size_t)(c >> 2) * N + p) * 128;
    int cc = (c & 3) * 32;
    Frag f;
    f.q[0] = *(const u32x4*)(row + cc + kb);
    f.q[1] = *(const u32x4*)(row + cc + 16 + kb);
#pragma unroll
    for (int t = 0; t < 8; ++t)
      acc[t] = wmma_f16(f.v, load_frag(po0 + ((c * 8 + t) * 32 + lane) * 16), acc[t]);
  }
#pragma unroll
  for (int t = 0; t < 8; ++t) store_tile(hb, 128, t, acc[t], ob0, lane, true);

  // GEMM2: 128->128; keep all 4 A fragments in regs, then overwrite hb.
  v16h af[4];
#pragma unroll
  for (int c = 0; c < 4; ++c) af[c] = frag_lds(hb, 128, c, lane);
#pragma unroll
  for (int t = 0; t < 8; ++t) {
    v8f a2 = zero;
#pragma unroll
    for (int c = 0; c < 4; ++c)
      a2 = wmma_f16(af[c], load_frag(po1 + ((c * 8 + t) * 32 + lane) * 16), a2);
    acc[t] = a2;
  }
#pragma unroll
  for (int t = 0; t < 8; ++t) store_tile(hb, 128, t, acc[t], ob1, lane, true);

  // GEMM3: 128->16 (one tile).
  v8f a3 = zero;
#pragma unroll
  for (int c = 0; c < 4; ++c) {
    v16h a = frag_lds(hb, 128, c, lane);
    a3 = wmma_f16(a, load_frag(po2 + (c * 32 + lane) * 16), a3);
  }
  int n = lane & 15, mb = (lane >> 4) * 8;
  float bv = ob2[n];
#pragma unroll
  for (int r = 0; r < 8; ++r) {
    float v = a3[r] + bv;
    int pp = base + mb + r;
    if (n < 8)
      out[(size_t)pp * 8 + n] = v;                              // mu
    else
      out[(size_t)N * 8 + (size_t)pp * 8 + (n - 8)] = __expf(v); // exp(sigma)
  }
}

// ---------------------------------------------------------------------------
extern "C" void kernel_launch(void* const* d_in, const int* in_sizes, int n_in,
                              void* d_out, int out_size, void* d_ws, size_t ws_size,
                              hipStream_t stream) {
  (void)in_sizes; (void)n_in; (void)out_size; (void)ws_size;
  const int N = 20000, K = 16;

  const float* inp = (const float*)d_in[0];
  const int* knn = (const int*)d_in[1];
  const float* W00 = (const float*)d_in[2];  const float* B00 = (const float*)d_in[3];
  const float* W01 = (const float*)d_in[4];  const float* B01 = (const float*)d_in[5];
  const float* W02 = (const float*)d_in[6];  const float* B02 = (const float*)d_in[7];
  const float* W10 = (const float*)d_in[8];  const float* B10 = (const float*)d_in[9];
  const float* W20 = (const float*)d_in[10]; const float* B20 = (const float*)d_in[11];
  const float* AW0 = (const float*)d_in[12]; const float* AB0 = (const float*)d_in[13];
  const float* AW1 = (const float*)d_in[14]; const float* AB1 = (const float*)d_in[15];
  const float* AW2 = (const float*)d_in[16]; const float* AB2 = (const float*)d_in[17];
  const float* OW0 = (const float*)d_in[18]; const float* OB0 = (const float*)d_in[19];
  const float* OW1 = (const float*)d_in[20]; const float* OB1 = (const float*)d_in[21];
  const float* OW2 = (const float*)d_in[22]; const float* OB2 = (const float*)d_in[23];

  // Workspace carve-up (~36 MB).
  float* x0f = (float*)d_ws;                       // N*128 f32 (residual master)
  float* x1f = x0f + (size_t)N * 128;              // N*128 f32
  _Float16* xh16 = (_Float16*)(x1f + (size_t)N * 128);  // 3 * N*128 f16
  _Float16* wp = xh16 + (size_t)3 * N * 128;       // packed weights (f16)
  _Float16* pg0a = wp;            // 12->32 : 1c x 2t  = 1024
  _Float16* pg0b = pg0a + 1024;   // 32->64 : 1c x 4t  = 2048
  _Float16* pg0c = pg0b + 2048;   // 64->128: 2c x 8t  = 8192
  _Float16* pg1  = pg0c + 8192;   // 137->128 (pad160): 5c x 8t = 20480
  _Float16* pg2  = pg1 + 20480;
  _Float16* pa0  = pg2 + 20480;   // attn packs: 4c x 1t = 2048 each
  _Float16* pa1  = pa0 + 2048;
  _Float16* pa2  = pa1 + 2048;
  _Float16* po0  = pa2 + 2048;    // 384->128: 12c x 8t = 49152
  _Float16* po1  = po0 + 49152;   // 128->128: 4c x 8t  = 16384
  _Float16* po2  = po1 + 16384;   // 128->16 : 4c x 1t  = 2048

  auto pack = [&](const float* W, _Float16* dst, int Kact, int Nact, int Nt,
                  int ch, int mode) {
    int total = ch * Nt * 512;
    pack_weight_kernel<<<dim3((total + 255) / 256), dim3(256), 0, stream>>>(
        W, dst, Kact, Nact, Nt, ch, mode, total);
  };
  pack(W00, pg0a, 12, 32, 2, 1, 0);
  pack(W01, pg0b, 32, 64, 4, 1, 0);
  pack(W02, pg0c, 64, 128, 8, 2, 0);
  pack(W10, pg1, 137, 128, 8, 5, 0);
  pack(W20, pg2, 137, 128, 8, 5, 0);
  pack(AW0, pa0, 128, 16, 1, 4, 1);
  pack(AW1, pa1, 128, 16, 1, 4, 1);
  pack(AW2, pa2, 128, 16, 1, 4, 1);
  pack(OW0, po0, 384, 128, 8, 12, 0);
  pack(OW1, po1, 128, 128, 8, 4, 0);
  pack(OW2, po2, 128, 16, 1, 4, 0);

  dim3 blk(128);                    // 4 waves / block
  dim3 grd((N + 3) / 4);            // 1 wave == 1 point
  graph_layer0<<<grd, blk, 0, stream>>>(inp, knn, pg0a, B00, pg0b, B01, pg0c, B02,
                                        pa0, AB0, x0f, xh16, N);
  graph_layer<<<grd, blk, 0, stream>>>(inp, knn + (size_t)N * K, xh16, x0f,
                                       pg1, B10, pa1, AB1, x1f,
                                       xh16 + (size_t)N * 128, N);
  graph_layer<<<grd, blk, 0, stream>>>(inp, knn + (size_t)2 * N * K,
                                       xh16 + (size_t)N * 128, x1f,
                                       pg2, B20, pa2, AB2, nullptr,
                                       xh16 + (size_t)2 * N * 128, N);
  head_kernel<<<dim3((N / 16 + 3) / 4), blk, 0, stream>>>(
      xh16, po0, OB0, po1, OB1, po2, OB2, (float*)d_out, N);
}
